// Hnn_80384607912348
// MI455X (gfx1250) — matrix operand order repaired and checked by reading, not compiled, not run on hardware
//
#include <hip/hip_runtime.h>
#include <hip/hip_bf16.h>
#include <stdint.h>

#define NUM_ENTS  100000
#define NUM_RELS2 1000
#define RPAD      1024     // relation rows padded to GEMM tile multiple
#define DIM       512
#define NUM_E     200000
#define RPB       32       // rows per GEMM block
#define LDSPITCH  (DIM + 8)   // +16B per row: de-conflict LDS banks

typedef __bf16 bf16_t;
typedef __attribute__((ext_vector_type(16))) __bf16 bf16x16;
typedef __attribute__((ext_vector_type(8)))  __bf16 bf16x8;
typedef __attribute__((ext_vector_type(4)))  __bf16 bf16x4;
typedef __attribute__((ext_vector_type(8)))  float  floatx8;

__device__ __forceinline__ bf16x16 pack16(const bf16_t* lo, const bf16_t* hi) {
    bf16x8 a = *(const bf16x8*)lo;   // 16B -> ds_load_b128 / global_load_b128
    bf16x8 b = *(const bf16x8*)hi;
    bf16x16 r;
#pragma unroll
    for (int h = 0; h < 8; ++h) { r[h] = a[h]; r[8 + h] = b[h]; }
    return r;
}

// async global->LDS 16B copy (CDNA5 async path, tracked by ASYNCcnt)
#define ASYNC16(ldsoff, gptr, imm)                                            \
    asm volatile("global_load_async_to_lds_b128 %0, %1, off offset:" #imm     \
                 :: "v"(ldsoff), "v"(gptr) : "memory")

// ---------------------------------------------------------------------------
// L2 row normalization: one wave per row; writes fp32 state + bf16 copy.
// ---------------------------------------------------------------------------
__global__ __launch_bounds__(256) void k_l2norm(const float* __restrict__ src,
                                                float* __restrict__ dst,
                                                bf16_t* __restrict__ dstbf, int N) {
    const int lane = threadIdx.x & 31;
    const int wave = threadIdx.x >> 5;
    const int row  = blockIdx.x * 8 + wave;
    if (row >= N) return;
    const float4* s = (const float4*)(src + (size_t)row * DIM);
    float4 v[4];
    float ss = 0.0f;
#pragma unroll
    for (int i = 0; i < 4; ++i) {
        v[i] = s[lane + 32 * i];
        ss += v[i].x * v[i].x + v[i].y * v[i].y + v[i].z * v[i].z + v[i].w * v[i].w;
    }
#pragma unroll
    for (int off = 16; off > 0; off >>= 1)
        ss += __shfl_xor(ss, off, 32);                  // wave32 reduction
    const float inv = 1.0f / fmaxf(sqrtf(ss), 1e-12f);
    float4* d = (float4*)(dst + (size_t)row * DIM);
#pragma unroll
    for (int i = 0; i < 4; ++i) {
        float4 o = v[i];
        o.x *= inv; o.y *= inv; o.z *= inv; o.w *= inv;
        d[lane + 32 * i] = o;
        if (dstbf) {
            bf16x4 ob;
            ob[0] = (bf16_t)o.x; ob[1] = (bf16_t)o.y;
            ob[2] = (bf16_t)o.z; ob[3] = (bf16_t)o.w;
            *(bf16x4*)(dstbf + (size_t)row * DIM + 4 * (lane + 32 * i)) = ob;
        }
    }
}

__global__ __launch_bounds__(256) void k_zero(float* __restrict__ p, long long n) {
    long long i = (long long)blockIdx.x * blockDim.x + threadIdx.x;
    if (i < n) p[i] = 0.0f;
}

// one-time fp32 -> bf16 weight conversion (row-major preserved)
__global__ __launch_bounds__(256) void k_f32_to_bf16(const float* __restrict__ src,
                                                     bf16_t* __restrict__ dst,
                                                     long long n) {
    long long i = ((long long)blockIdx.x * blockDim.x + threadIdx.x) * 8;
    if (i >= n) return;
    float4 a = ((const float4*)(src + i))[0];
    float4 b = ((const float4*)(src + i))[1];
    bf16x8 o;
    o[0] = (bf16_t)a.x; o[1] = (bf16_t)a.y; o[2] = (bf16_t)a.z; o[3] = (bf16_t)a.w;
    o[4] = (bf16_t)b.x; o[5] = (bf16_t)b.y; o[6] = (bf16_t)b.z; o[7] = (bf16_t)b.w;
    *(bf16x8*)(dst + i) = o;
}

// ---------------------------------------------------------------------------
// Fused gather + segment-sum: one block per edge, 128 lanes x float4.
// ---------------------------------------------------------------------------
__global__ __launch_bounds__(128) void k_scatter_acc(const float* __restrict__ src,
                                                     const int* __restrict__ srcIdx,
                                                     const int* __restrict__ dstIdx,
                                                     float* __restrict__ acc,
                                                     float* __restrict__ cnt, int E) {
    const int e = blockIdx.x;
    if (e >= E) return;
    const int si = srcIdx[e];
    const int di = dstIdx[e];
    const float4 v = ((const float4*)(src + (size_t)si * DIM))[threadIdx.x];
    float* a = acc + (size_t)di * DIM + threadIdx.x * 4;
    atomicAdd(a + 0, v.x);
    atomicAdd(a + 1, v.y);
    atomicAdd(a + 2, v.z);
    atomicAdd(a + 3, v.w);
    if (threadIdx.x == 0) atomicAdd(cnt + di, 1.0f);
}

// mean = acc / max(cnt,1), emitted directly as bf16 GEMM operand
__global__ __launch_bounds__(256) void k_mean_div_bf(const float* __restrict__ acc,
                                                     const float* __restrict__ cnt,
                                                     bf16_t* __restrict__ outbf,
                                                     long long n) {
    long long i = ((long long)blockIdx.x * blockDim.x + threadIdx.x) * 8;
    if (i >= n) return;
    const float r = 1.0f / fmaxf(cnt[i / DIM], 1.0f);
    float4 a = ((const float4*)(acc + i))[0];
    float4 b = ((const float4*)(acc + i))[1];
    bf16x8 o;
    o[0] = (bf16_t)(a.x * r); o[1] = (bf16_t)(a.y * r);
    o[2] = (bf16_t)(a.z * r); o[3] = (bf16_t)(a.w * r);
    o[4] = (bf16_t)(b.x * r); o[5] = (bf16_t)(b.y * r);
    o[6] = (bf16_t)(b.z * r); o[7] = (bf16_t)(b.w * r);
    *(bf16x8*)(outbf + i) = o;
}

// ---------------------------------------------------------------------------
// Fused dual-GEMM update (all-bf16 operands, fp32 accumulate):
//   out[r,:] = cnt[r]>0 ? 0.5*relu(M@Wmsg^T+bmsg) + 0.5*(X@Wself^T+bself) : X[r,:]
// Block = 256 thr = 8 waves; tile 32 rows x 512 cols; wave = 16 rows x 128 cols.
// A slabs staged in LDS (async-to-LDS); B fragments software-pipelined 1 deep
// so loads for subtile t+1 are in flight while WMMA t executes.
// ---------------------------------------------------------------------------
__global__ __launch_bounds__(256) void k_fused_update(
    const float* __restrict__ X,          // fp32 states (fallback copy)
    const bf16_t* __restrict__ Xbf,       // [Npad, DIM]
    const bf16_t* __restrict__ Mbf,       // [Npad, DIM]
    const bf16_t* __restrict__ Wmsg,      // [DIM, DIM] bf16 row-major
    const float*  __restrict__ bmsg,
    const bf16_t* __restrict__ Wself,
    const float*  __restrict__ bself,
    const float*  __restrict__ cnt,
    float* __restrict__ out, int N) {
    __shared__ __align__(16) bf16_t sM[RPB * LDSPITCH];
    __shared__ __align__(16) bf16_t sX[RPB * LDSPITCH];

    const int tid    = threadIdx.x;
    const int lane   = tid & 31;
    const int wave   = tid >> 5;
    const int rowBlk = blockIdx.x * RPB;

    // ---- async copy A slabs: 32 rows x 1KB; 8 threads/row, 128B/thread ----
    {
        const int r   = tid >> 3;          // 0..31
        const int seg = tid & 7;           // 0..7, 128B segments
        const bf16_t* gM = Mbf + (size_t)(rowBlk + r) * DIM + seg * 64;
        const bf16_t* gX = Xbf + (size_t)(rowBlk + r) * DIM + seg * 64;
        const uint32_t lM = (uint32_t)(uintptr_t)(void*)sM + (uint32_t)(r * LDSPITCH + seg * 64) * 2u;
        const uint32_t lX = (uint32_t)(uintptr_t)(void*)sX + (uint32_t)(r * LDSPITCH + seg * 64) * 2u;
        ASYNC16(lM, gM, 0);   ASYNC16(lM, gM, 16);
        ASYNC16(lM, gM, 32);  ASYNC16(lM, gM, 48);
        ASYNC16(lM, gM, 64);  ASYNC16(lM, gM, 80);
        ASYNC16(lM, gM, 96);  ASYNC16(lM, gM, 112);
        ASYNC16(lX, gX, 0);   ASYNC16(lX, gX, 16);
        ASYNC16(lX, gX, 32);  ASYNC16(lX, gX, 48);
        ASYNC16(lX, gX, 64);  ASYNC16(lX, gX, 80);
        ASYNC16(lX, gX, 96);  ASYNC16(lX, gX, 112);
        asm volatile("s_wait_asynccnt 0" ::: "memory");
    }
    __syncthreads();

    const int rowHalf = wave >> 2;         // 0/1 -> which 16-row group
    const int colGrp  = wave & 3;          // 0..3 -> 128-col group
    const int halfSel = lane >> 4;
    const int l15     = lane & 15;
    const int aRow    = (rowHalf * 16 + l15) * LDSPITCH;

    floatx8 accM[8] = {};
    floatx8 accS[8] = {};

    // fragment loaders
    // A frag (16x32 bf16): lanes 0-15 K={0..7,16..23}, lanes 16-31 K={8..15,24..31}
    #define LDA(s, k0) pack16(&(s)[aRow + (k0) + (halfSel ? 8 : 0)], \
                              &(s)[aRow + (k0) + (halfSel ? 8 : 0) + 16])
    #define LDB(W, t, k0) pack16((W) + (size_t)(colGrp * 128 + (t) * 16 + l15) * DIM + (k0) + halfSel * 8, \
                                 (W) + (size_t)(colGrp * 128 + (t) * 16 + l15) * DIM + (k0) + halfSel * 8 + 16)

    // prime the B pipeline (t=0, k0=0)
    bf16x16 bM = LDB(Wmsg, 0, 0);
    bf16x16 bS = LDB(Wself, 0, 0);

    for (int k0 = 0; k0 < DIM; k0 += 32) {
        const bf16x16 aM = LDA(sM, k0);
        const bf16x16 aS = LDA(sX, k0);
        const int kNext = (k0 + 32 < DIM) ? (k0 + 32) : k0;   // clamped, branch-free
#pragma unroll
        for (int t = 0; t < 8; ++t) {
            const int tn = (t + 1) & 7;
            const int kb = (t == 7) ? kNext : k0;
            const bf16x16 bMn = LDB(Wmsg, tn, kb);            // in flight during wmma t
            const bf16x16 bSn = LDB(Wself, tn, kb);
            accM[t] = __builtin_amdgcn_wmma_f32_16x16x32_bf16(
                false, aM, false, bM, (short)0, accM[t], false, false);
            accS[t] = __builtin_amdgcn_wmma_f32_16x16x32_bf16(
                false, aS, false, bS, (short)0, accS[t], false, false);
            bM = bMn; bS = bSn;
        }
    }
    #undef LDA
    #undef LDB

    // Epilogue. C/D: VGPR v, lanes 0-15 -> row v; lanes 16-31 -> row v+8.
#pragma unroll
    for (int t = 0; t < 8; ++t) {
        const int col  = colGrp * 128 + t * 16 + l15;
        const float bm = bmsg[col];
        const float bs = bself[col];
#pragma unroll
        for (int v = 0; v < 8; ++v) {
            const int row = rowBlk + rowHalf * 16 + v + halfSel * 8;
            if (row < N) {
                float msg = accM[t][v] + bm;
                msg = msg > 0.0f ? msg : 0.0f;
                const float val = 0.5f * msg + 0.5f * (accS[t][v] + bs);
                const size_t oi = (size_t)row * DIM + col;
                out[oi] = (cnt[row] > 0.0f) ? val : X[oi];
            }
        }
    }
}

// ---------------------------------------------------------------------------
// Host orchestration (two layers unrolled, graph-capture safe).
// ---------------------------------------------------------------------------
static inline void launch_zero(float* p, long long n, hipStream_t s) {
    k_zero<<<(unsigned)((n + 255) / 256), 256, 0, s>>>(p, n);
}

extern "C" void kernel_launch(void* const* d_in, const int* in_sizes, int n_in,
                              void* d_out, int out_size, void* d_ws, size_t ws_size,
                              hipStream_t stream) {
    const float* ent_emb = (const float*)d_in[0];
    const float* rel_emb = (const float*)d_in[1];
    const float* w_r_w   = (const float*)d_in[2];
    const float* w_r_b   = (const float*)d_in[3];
    const float* w_u2r_w = (const float*)d_in[4];
    const float* w_u2r_b = (const float*)d_in[5];
    const float* w_u_w   = (const float*)d_in[6];
    const float* w_u_b   = (const float*)d_in[7];
    const float* w_r2u_w = (const float*)d_in[8];
    const float* w_r2u_b = (const float*)d_in[9];
    const int*   idx_ent = (const int*)d_in[10];
    const int*   idx_rel = (const int*)d_in[11];
    (void)in_sizes; (void)n_in; (void)out_size; (void)ws_size;

    const size_t ND = (size_t)NUM_ENTS * DIM;
    const size_t RD = (size_t)NUM_RELS2 * DIM;
    const size_t WD = (size_t)DIM * DIM;

    float* ws = (float*)d_ws;
    size_t o = 0;
    float* g_h   = ws + o; o += ND;
    float* tmp_u = ws + o; o += ND;
    float* acc_u = ws + o; o += ND;
    float* cnt_u = ws + o; o += NUM_ENTS;
    float* g_h0  = ws + o; o += RD;
    float* tmp_r = ws + o; o += RD;
    float* acc_r = ws + o; o += RD;
    float* cnt_r = ws + o; o += NUM_RELS2;
    // bf16 region (sizes in float units = elems/2; all counts multiple of 256)
    bf16_t* xbf_u  = (bf16_t*)(ws + o); o += ND / 2;                 // bf16(g_h)
    bf16_t* mbf_u  = (bf16_t*)(ws + o); o += ND / 2;                 // bf16 mean
    bf16_t* xbf_r  = (bf16_t*)(ws + o); o += (size_t)RPAD * DIM / 2;
    bf16_t* mbf_r  = (bf16_t*)(ws + o); o += (size_t)RPAD * DIM / 2;
    bf16_t* wbf_r   = (bf16_t*)(ws + o); o += WD / 2;
    bf16_t* wbf_u2r = (bf16_t*)(ws + o); o += WD / 2;
    bf16_t* wbf_u   = (bf16_t*)(ws + o); o += WD / 2;
    bf16_t* wbf_r2u = (bf16_t*)(ws + o); o += WD / 2;

    const unsigned wGrid = (unsigned)(WD / 8 / 256);
    k_f32_to_bf16<<<wGrid, 256, 0, stream>>>(w_r_w,   wbf_r,   (long long)WD);
    k_f32_to_bf16<<<wGrid, 256, 0, stream>>>(w_u2r_w, wbf_u2r, (long long)WD);
    k_f32_to_bf16<<<wGrid, 256, 0, stream>>>(w_u_w,   wbf_u,   (long long)WD);
    k_f32_to_bf16<<<wGrid, 256, 0, stream>>>(w_r2u_w, wbf_r2u, (long long)WD);

    const int entNormGrid = (NUM_ENTS + 7) / 8;
    const int relNormGrid = (NUM_RELS2 + 7) / 8;
    const int entUpdGrid  = NUM_ENTS / RPB;                  // 100000/32 = 3125
    const int relUpdGrid  = (NUM_RELS2 + RPB - 1) / RPB;     // 32 (reads pad rows)

    k_l2norm<<<entNormGrid, 256, 0, stream>>>(ent_emb, g_h, xbf_u, NUM_ENTS);
    k_l2norm<<<relNormGrid, 256, 0, stream>>>(rel_emb, g_h0, xbf_r, NUM_RELS2);

    for (int layer = 0; layer < 2; ++layer) {
        // ---- entity -> relation ----
        launch_zero(acc_r, (long long)RD, stream);
        launch_zero(cnt_r, NUM_RELS2, stream);
        k_scatter_acc<<<NUM_E, 128, 0, stream>>>(g_h, idx_ent, idx_rel,
                                                 acc_r, cnt_r, NUM_E);
        k_mean_div_bf<<<(unsigned)(RD / 8 / 256), 256, 0, stream>>>(
            acc_r, cnt_r, mbf_r, (long long)RD);
        k_fused_update<<<relUpdGrid, 256, 0, stream>>>(
            g_h0, xbf_r, mbf_r, wbf_u2r, w_u2r_b, wbf_r, w_r_b, cnt_r,
            tmp_r, NUM_RELS2);
        // ---- relation (updated, un-normalized) -> entity ----
        launch_zero(acc_u, (long long)ND, stream);
        launch_zero(cnt_u, NUM_ENTS, stream);
        k_scatter_acc<<<NUM_E, 128, 0, stream>>>(tmp_r, idx_rel, idx_ent,
                                                 acc_u, cnt_u, NUM_E);
        k_mean_div_bf<<<(unsigned)(ND / 8 / 256), 256, 0, stream>>>(
            acc_u, cnt_u, mbf_u, (long long)ND);
        k_fused_update<<<entUpdGrid, 256, 0, stream>>>(
            g_h, xbf_u, mbf_u, wbf_r2u, w_r2u_b, wbf_u, w_u_b, cnt_u,
            tmp_u, NUM_ENTS);
        // ---- renormalize (final layer writes d_out; bf copies feed layer 2) ----
        float* entDst = (layer == 1) ? (float*)d_out : g_h;
        k_l2norm<<<entNormGrid, 256, 0, stream>>>(tmp_u, entDst, xbf_u, NUM_ENTS);
        k_l2norm<<<relNormGrid, 256, 0, stream>>>(tmp_r, g_h0, xbf_r, NUM_RELS2);
    }
}
